// GATLayer_86148454023362
// MI455X (gfx1250) — matrix operand order repaired and checked by reading, not compiled, not run on hardware
//
#include <hip/hip_runtime.h>
#include <hip/hip_bf16.h>

typedef __attribute__((ext_vector_type(16))) _Float16 v16h;
typedef __attribute__((ext_vector_type(8)))  _Float16 v8h;
typedef __attribute__((ext_vector_type(8)))  float    v8f;

#define IN_F   256
#define HEADS  4
#define F_OUT  32
#define OUT_C  128      // HEADS*F_OUT
#define BCOLS  144      // 128 (W_mlp) + 4 (W1) + 4 (W2) + 8 pad -> 9 col tiles of 16

// ---------------------------------------------------------------------------
// Prep: pack B = [W_mlp cols | W1 rows | W2 rows | zero pad] column-major f16
// (each column's 256 K-values contiguous) + combined bias vector (144 floats).
// grid = 144 blocks (one per column), block = 256 threads (one per K).
// ---------------------------------------------------------------------------
__global__ __launch_bounds__(256)
void gat_prep_kernel(const float* __restrict__ W_mlp, const float* __restrict__ b_mlp,
                     const float* __restrict__ W1,    const float* __restrict__ b1,
                     const float* __restrict__ W2,    const float* __restrict__ b2,
                     _Float16* __restrict__ Bt, float* __restrict__ biasC) {
    const int j = blockIdx.x;     // column 0..143
    const int k = threadIdx.x;    // K 0..255
    float val, bv;
    if (j < OUT_C)            { val = W_mlp[k * OUT_C + j];        bv = b_mlp[j]; }
    else if (j < OUT_C + 4)   { val = W1[(j - OUT_C) * IN_F + k];  bv = b1[j - OUT_C]; }
    else if (j < OUT_C + 8)   { val = W2[(j - OUT_C - 4) * IN_F + k]; bv = b2[j - OUT_C - 4]; }
    else                      { val = 0.0f; bv = 0.0f; }
    Bt[j * IN_F + k] = (_Float16)val;
    if (k == 0) biasC[j] = bv;
}

// ---------------------------------------------------------------------------
// Fused WMMA GEMM: [16 x 256] x [256 x 144] per block.
// 9 waves: wave 0..7 -> X[N,128] and out = eps*X ; wave 8 -> H12[N,8] (h1|h2).
// A tile staged in LDS as f16; K-loop of 8 x v_wmma_f32_16x16x32_f16.
// grid = N/16 blocks, block = 288 threads (9 waves of 32).
// ---------------------------------------------------------------------------
__global__ __launch_bounds__(288)
void gat_gemm_kernel(const float* __restrict__ feat,
                     const _Float16* __restrict__ Bt,   // [144][256] col-major
                     const float* __restrict__ biasC,   // [144]
                     const float* __restrict__ eps,     // [1]
                     float* __restrict__ X,             // [N][128]
                     float* __restrict__ H12,           // [N][8] = h1(4)|h2(4)
                     float* __restrict__ out) {         // [N][128]
    __shared__ _Float16 sA[16 * IN_F];                  // 8 KB

    const int tid = threadIdx.x;
    const int rowBase = blockIdx.x * 16;

    // Stage A tile: 16 rows x 256 f32 -> f16 LDS (float4 granularity).
    for (int idx = tid; idx < 16 * 64; idx += 288) {
        const int r  = idx >> 6;        // row 0..15
        const int c4 = idx & 63;        // float4 index 0..63
        const float4 f = *(const float4*)(feat + (size_t)(rowBase + r) * IN_F + c4 * 4);
        _Float16* p = &sA[r * IN_F + c4 * 4];
        p[0] = (_Float16)f.x; p[1] = (_Float16)f.y;
        p[2] = (_Float16)f.z; p[3] = (_Float16)f.w;
    }
    __syncthreads();

    const int wave = tid >> 5;          // 0..8 (column tile)
    const int lane = tid & 31;
    const int ln   = lane & 15;
    const int hi   = lane >> 4;
    const int col  = wave * 16 + ln;    // 0..143

    v8f acc = {};
    const _Float16* bp = Bt + (size_t)col * IN_F;

    #pragma unroll
    for (int kt = 0; kt < 8; ++kt) {
        union { v16h v; v8h h[2]; } a, b;
        // A lane layout: M = ln; K halves {8*hi .. 8*hi+7} and {+16}
        const int ka = kt * 32 + hi * 8;
        a.h[0] = *(const v8h*)&sA[ln * IN_F + ka];
        a.h[1] = *(const v8h*)&sA[ln * IN_F + ka + 16];
        // B lane layout: N = ln; K halves {8*hi .. 8*hi+7} and {+16} (col-major mem)
        const int kb = kt * 32 + hi * 8;
        b.h[0] = *(const v8h*)(bp + kb);
        b.h[1] = *(const v8h*)(bp + kb + 16);
        acc = __builtin_amdgcn_wmma_f32_16x16x32_f16(
            /*neg_a=*/false, a.v, /*neg_b=*/false, b.v,
            /*c_mod=*/(short)0, acc, /*reuse_a=*/false, /*reuse_b=*/false);
    }

    const float bias = biasC[col];
    const float e = eps[0];

    if (wave < 8) {
        // C/D layout: element r -> row (r + 8*hi), column ln
        #pragma unroll
        for (int r = 0; r < 8; ++r) {
            const int row = rowBase + r + 8 * hi;
            const float val = acc[r] + bias;
            X[(size_t)row * OUT_C + col]   = val;
            out[(size_t)row * OUT_C + col] = e * val;   // init: agg added later
        }
    } else if (ln < 8) {
        #pragma unroll
        for (int r = 0; r < 8; ++r) {
            const int row = rowBase + r + 8 * hi;
            H12[(size_t)row * 8 + ln] = acc[r] + bias;  // ln<4: h1, ln>=4: h2
        }
    }
}

// ---------------------------------------------------------------------------
// Edge scatter: one wave32 per edge. lane t handles 4 features [4t,4t+4),
// head h = t>>3. v = relu(h1[src,h]+h2[dst,h]); out[src] += v * X[dst].
// grid = E/8, block = 256 (8 edges per block).
// ---------------------------------------------------------------------------
__global__ __launch_bounds__(256)
void gat_edge_kernel(const int* __restrict__ ei,   // [2][E]
                     const float* __restrict__ X,
                     const float* __restrict__ H12,
                     float* __restrict__ out, int E) {
    const int t = threadIdx.x & 31;
    const int e = blockIdx.x * 8 + (threadIdx.x >> 5);
    if (e >= E) return;

    const int src = ei[e];
    const int dst = ei[E + e];
    const int h = t >> 3;

    float v = H12[(size_t)src * 8 + h] + H12[(size_t)dst * 8 + 4 + h];
    v = v > 0.0f ? v : 0.0f;

    const float4 xv = *(const float4*)(X + (size_t)dst * OUT_C + t * 4);
    float* op = out + (size_t)src * OUT_C + t * 4;
    __hip_atomic_fetch_add(op + 0, xv.x * v, __ATOMIC_RELAXED, __HIP_MEMORY_SCOPE_AGENT);
    __hip_atomic_fetch_add(op + 1, xv.y * v, __ATOMIC_RELAXED, __HIP_MEMORY_SCOPE_AGENT);
    __hip_atomic_fetch_add(op + 2, xv.z * v, __ATOMIC_RELAXED, __HIP_MEMORY_SCOPE_AGENT);
    __hip_atomic_fetch_add(op + 3, xv.w * v, __ATOMIC_RELAXED, __HIP_MEMORY_SCOPE_AGENT);
}

// ---------------------------------------------------------------------------
extern "C" void kernel_launch(void* const* d_in, const int* in_sizes, int n_in,
                              void* d_out, int out_size, void* d_ws, size_t ws_size,
                              hipStream_t stream) {
    const float* feat  = (const float*)d_in[0];
    const int*   eidx  = (const int*)d_in[1];
    const float* W_mlp = (const float*)d_in[2];
    const float* b_mlp = (const float*)d_in[3];
    const float* W1    = (const float*)d_in[4];
    const float* b1    = (const float*)d_in[5];
    const float* W2    = (const float*)d_in[6];
    const float* b2    = (const float*)d_in[7];
    const float* eps   = (const float*)d_in[8];

    const int N = in_sizes[0] / IN_F;   // 50000
    const int E = in_sizes[1] / 2;      // 800000

    // Workspace layout
    char* ws = (char*)d_ws;
    _Float16* Bt   = (_Float16*)ws;                         ws += (size_t)BCOLS * IN_F * sizeof(_Float16); // 73.7 KB
    float*    biasC = (float*)ws;                           ws += (size_t)BCOLS * sizeof(float);
    ws = (char*)(((uintptr_t)ws + 255) & ~(uintptr_t)255);
    float*    X    = (float*)ws;                            ws += (size_t)N * OUT_C * sizeof(float);       // 25.6 MB
    float*    H12  = (float*)ws;                            /* N*8 floats: 1.6 MB */

    float* out = (float*)d_out;

    gat_prep_kernel<<<BCOLS, IN_F, 0, stream>>>(W_mlp, b_mlp, W1, b1, W2, b2, Bt, biasC);

    gat_gemm_kernel<<<N / 16, 288, 0, stream>>>(feat, Bt, biasC, eps, X, H12, out);

    gat_edge_kernel<<<(E + 7) / 8, 256, 0, stream>>>(eidx, X, H12, out, E);
}
